// ChannelMultiHeadAttention_27075473834612
// MI455X (gfx1250) — matrix operand order repaired and checked by reading, not compiled, not run on hardware
//
#include <hip/hip_runtime.h>

#define SEQ    14
#define EMB    64
#define BATCH  1024
#define CHAN   64
#define WAVES  8
#define TPW    4
#define LN_EPS 1e-5f

typedef __attribute__((ext_vector_type(16))) __bf16 v16bf;
typedef __attribute__((ext_vector_type(8)))  __bf16 v8bf;
typedef __attribute__((ext_vector_type(8)))  float  v8f;
typedef __attribute__((ext_vector_type(4)))  unsigned int u32x4;
typedef __attribute__((ext_vector_type(8)))  int   i32x8;
typedef __attribute__((ext_vector_type(4)))  int   i32x4;

static __device__ __forceinline__ v8f wmma_bf16(v16bf a, v16bf b, v8f c) {
  // D = A(16x32 bf16) x B(32x16 bf16) + C(16x16 f32)
  return __builtin_amdgcn_wmma_f32_16x16x32_bf16(false, a, false, b, (short)0, c, false, false);
}

// ---- WMMA fragment builders (wave32 layouts per CDNA5 ISA) ----
// A (16x32): lane m = lane&15, half = lane>>4; element e -> k = ((e&8)<<1)+8*half+(e&7)
static __device__ __forceinline__ v16bf afrag_bf16(const __bf16* p /*&arr[m][kc*32]*/, int half) {
  v8bf lo = *(const v8bf*)(p + half * 8);
  v8bf hi = *(const v8bf*)(p + 16 + half * 8);
  v16bf a;
#pragma unroll
  for (int i = 0; i < 8; ++i) { a[i] = lo[i]; a[i + 8] = hi[i]; }
  return a;
}
static __device__ __forceinline__ v16bf afrag_f32(const float* p /*&arr[m][kc*32]*/, int half) {
  v8f lo = *(const v8f*)(p + half * 8);
  v8f hi = *(const v8f*)(p + 16 + half * 8);
  v16bf a;
#pragma unroll
  for (int i = 0; i < 8; ++i) { a[i] = (__bf16)lo[i]; a[i + 8] = (__bf16)hi[i]; }
  return a;
}
// B (32x16): lane n = lane&15; element e -> k = ((e&8)<<1)+8*half+(e&7)
static __device__ __forceinline__ v16bf bfrag(const __bf16* p /*&arr[n][kc*32]*/, int half) {
  v8bf lo = *(const v8bf*)(p + half * 8);
  v8bf hi = *(const v8bf*)(p + 16 + half * 8);
  v16bf b;
#pragma unroll
  for (int i = 0; i < 8; ++i) { b[i] = lo[i]; b[i + 8] = hi[i]; }
  return b;
}

// ---- Tensor Data Mover: 1-D f32 tile descriptor (D# groups 0/1, ISA 8.3/8.4) ----
static __device__ __forceinline__ void tdm_desc_1d(unsigned long long gaddr, unsigned lds_off,
                                                   unsigned nelem, u32x4* g0, i32x8* g1) {
  u32x4 a; i32x8 b;
  a[0] = 1u;                                                    // count=1, user mode
  a[1] = lds_off;                                               // lds_addr (bytes)
  a[2] = (unsigned)gaddr;                                       // global_addr[31:0]
  a[3] = (unsigned)((gaddr >> 32) & 0x01FFFFFFull) | (2u << 30);// global_addr[56:32] | type=2
  const unsigned td0 = 0x40000000u;                             // huge tensor_dim0: no OOB clip
  b[0] = (int)0x00020000u;                // workgroup_mask=0, data_size=2 (4 bytes)
  b[1] = (int)((td0 & 0xFFFFu) << 16);    // tensor_dim0[15:0]
  b[2] = (int)((td0 >> 16) & 0xFFFFu);    // tensor_dim0[31:16], tensor_dim1=0
  b[3] = (int)(nelem << 16);              // tile_dim0 = nelem (f32 elements)
  b[4] = 0;                               // tile_dim1=0 (1-D), tile_dim2=0
  b[5] = (int)nelem;                      // tensor_dim0_stride (unused for 1-D tile)
  b[6] = 0;
  b[7] = 0;
  *g0 = a; *g1 = b;
}
static __device__ __forceinline__ void tdm_load(u32x4 g0, i32x8 g1) {
  const i32x4 z4 = {0, 0, 0, 0};
#if defined(__clang_major__) && (__clang_major__ >= 23)
  const i32x8 z8 = {0, 0, 0, 0, 0, 0, 0, 0};
  __builtin_amdgcn_tensor_load_to_lds(g0, g1, z4, z4, z8, 0);
#else
  __builtin_amdgcn_tensor_load_to_lds(g0, g1, z4, z4, 0);
#endif
}
static __device__ __forceinline__ void tdm_store(u32x4 g0, i32x8 g1) {
  const i32x4 z4 = {0, 0, 0, 0};
#if defined(__clang_major__) && (__clang_major__ >= 23)
  const i32x8 z8 = {0, 0, 0, 0, 0, 0, 0, 0};
  __builtin_amdgcn_tensor_store_from_lds(g0, g1, z4, z4, z8, 0);
#else
  __builtin_amdgcn_tensor_store_from_lds(g0, g1, z4, z4, 0);
#endif
}
static __device__ __forceinline__ unsigned lds_off_of(const void* p) {
  // flat LDS aperture: addr[31:0] == byte offset within workgroup LDS
  return (unsigned)(unsigned long long)(uintptr_t)p;
}

struct WaveSlab {
  float  xf[2][16][EMB];  // TDM-loaded X (f32), double buffered; rows 14,15 zeroed once
  float  os[SEQ * EMB];   // output staging for TDM store
  __bf16 qs[16][EMB];     // Q rows -> A-source for energy
  __bf16 ks[16][EMB];     // K rows [kk][d] -> B-source for energy
  __bf16 vt[EMB][32];     // V transposed [e][l], K-padded to 32 -> B-source for out
  __bf16 et[16][32];      // energy transposed [v][l], K-padded -> B-source for mix
  __bf16 as_[16][32];     // att rows [a][l], K-padded -> A-source for out
  float  es[16][16];      // energy1 f32 staging for LN/softmax
};

__global__ __launch_bounds__(256) void cmha_kernel(
    const float* __restrict__ x,   const float* __restrict__ w_c,
    const float* __restrict__ Wq,  const float* __restrict__ bq,
    const float* __restrict__ Wk,  const float* __restrict__ bk,
    const float* __restrict__ Wv,  const float* __restrict__ bv,
    const float* __restrict__ lng, const float* __restrict__ lnb,
    float* __restrict__ out)
{
  __shared__ __bf16 wb[3][EMB * EMB];   // W row-major bf16: B[k=d][n=e] = wb[e*64+d]
  __shared__ float  bias_s[3][EMB];
  __shared__ float  wc_s[SEQ * SEQ];
  __shared__ float  lng_s[SEQ], lnb_s[SEQ];
  __shared__ WaveSlab slab[WAVES];

  const int tid  = threadIdx.x;
  const int wid  = tid >> 5;
  const int lane = tid & 31;
  const int half = lane >> 4;
  const int m16  = lane & 15;

  // ---- stage weights / constants (block-cooperative, once) ----
  const float* Ws[3] = {Wq, Wk, Wv};
  const float* Bs[3] = {bq, bk, bv};
#pragma unroll
  for (int p = 0; p < 3; ++p) {
    for (int i = tid; i < EMB * EMB; i += 256) wb[p][i] = (__bf16)Ws[p][i];
    if (tid < EMB) bias_s[p][tid] = Bs[p][tid];
  }
  for (int i = tid; i < SEQ * SEQ; i += 256) wc_s[i] = w_c[i];
  if (tid < SEQ) { lng_s[tid] = lng[tid]; lnb_s[tid] = lnb[tid]; }

  WaveSlab& S = slab[wid];
  // zero pad regions once (TDM / tile loop never write them; keeps pad lanes finite)
  for (int i = lane; i < 2 * 2 * EMB; i += 32)
    S.xf[i >> 7][14 + ((i >> 6) & 1)][i & 63] = 0.0f;
  for (int i = lane; i < EMB * 16; i += 32) S.vt[i >> 4][16 + (i & 15)] = (__bf16)0.0f;
  for (int i = lane; i < 16 * 16; i += 32) {
    S.et [i >> 4][16 + (i & 15)] = (__bf16)0.0f;
    S.as_[i >> 4][16 + (i & 15)] = (__bf16)0.0f;
  }
  for (int i = lane; i < 2 * EMB; i += 32) S.ks[14 + (i >> 6)][i & 63] = (__bf16)0.0f;
  __syncthreads();

  // w_c A-fragment (16x32, zero beyond 14 in both m and k), built once
  v16bf awc;
#pragma unroll
  for (int e = 0; e < 16; ++e) {
    const int k = ((e & 8) << 1) + half * 8 + (e & 7);
    const float v = (m16 < SEQ && k < SEQ) ? wc_s[m16 * SEQ + k] : 0.0f;
    awc[e] = (__bf16)v;
  }

  const int tile0 = (blockIdx.x * WAVES + wid) * TPW;

  // ---- prologue: TDM-load tile0 into xf[0] ----
  {
    u32x4 g0; i32x8 g1;
    tdm_desc_1d((unsigned long long)(uintptr_t)(x + (size_t)tile0 * (SEQ * EMB)),
                lds_off_of(&S.xf[0][0][0]), SEQ * EMB, &g0, &g1);
    tdm_load(g0, g1);
  }

  for (int t = 0; t < TPW; ++t) {
    const int tile = tile0 + t;

    // issue next tile's TDM load (double buffer) before waiting on current
    if (t + 1 < TPW) {
      u32x4 g0; i32x8 g1;
      tdm_desc_1d((unsigned long long)(uintptr_t)(x + (size_t)(tile + 1) * (SEQ * EMB)),
                  lds_off_of(&S.xf[(t + 1) & 1][0][0]), SEQ * EMB, &g0, &g1);
      tdm_load(g0, g1);
    }
    // TENSORcnt is in-order per wave: "<=1" proves load(t) AND store(t-1) completed
    asm volatile("" ::: "memory");
    if (t + 1 < TPW) __builtin_amdgcn_s_wait_tensorcnt(1);
    else             __builtin_amdgcn_s_wait_tensorcnt(0);
    asm volatile("" ::: "memory");

    const float* xt = &S.xf[t & 1][0][0];
    v16bf ax0 = afrag_f32(xt + m16 * EMB + 0,  half);
    v16bf ax1 = afrag_f32(xt + m16 * EMB + 32, half);

    // ---- Q,K,V projections: y = x @ W^T + b  (24 WMMA) ----
#pragma unroll
    for (int p = 0; p < 3; ++p) {
#pragma unroll
      for (int nt = 0; nt < 4; ++nt) {
        const int n = nt * 16 + m16;
        v16bf b0 = bfrag(&wb[p][n * EMB + 0],  half);
        v16bf b1 = bfrag(&wb[p][n * EMB + 32], half);
        v8f acc = {};
        acc = wmma_bf16(ax0, b0, acc);
        acc = wmma_bf16(ax1, b1, acc);
        const float bias = bias_s[p][n];
#pragma unroll
        for (int r = 0; r < 8; ++r) {
          const int mm = r + 8 * half;                 // C-layout row
          const __bf16 hv = (__bf16)(acc[r] + bias);
          if (p == 0)      S.qs[mm][n] = hv;
          else if (p == 1) S.ks[mm][n] = hv;
          else             S.vt[n][mm] = hv;           // V transposed
        }
      }
    }

    // ---- energy = Q @ K^T (2 WMMA) ----
    v16bf aq0 = afrag_bf16(&S.qs[m16][0],  half);
    v16bf aq1 = afrag_bf16(&S.qs[m16][32], half);
    v16bf bk0 = bfrag(&S.ks[m16][0],  half);
    v16bf bk1 = bfrag(&S.ks[m16][32], half);
    v8f en = {};
    en = wmma_bf16(aq0, bk0, en);
    en = wmma_bf16(aq1, bk1, en);
#pragma unroll
    for (int r = 0; r < 8; ++r) S.et[m16][r + 8 * half] = (__bf16)en[r]; // transposed store

    // ---- energy1 = w_c @ energy (1 WMMA, K padded to 32) ----
    v16bf be = bfrag(&S.et[m16][0], half);
    v8f e1 = {};
    e1 = wmma_bf16(awc, be, e1);
#pragma unroll
    for (int r = 0; r < 8; ++r) S.es[r + 8 * half][m16] = e1[r];

    // ---- LayerNorm(14) + softmax: one row per lane (lanes 0..15), VALU only ----
    if (half == 0) {
      const int row = m16;
      float vals[SEQ];
      float mu = 0.0f;
#pragma unroll
      for (int v = 0; v < SEQ; ++v) { vals[v] = S.es[row][v]; mu += vals[v]; }
      mu *= (1.0f / SEQ);
      float var = 0.0f;
#pragma unroll
      for (int v = 0; v < SEQ; ++v) { const float d = vals[v] - mu; var += d * d; }
      var *= (1.0f / SEQ);
      const float rstd = __frsqrt_rn(var + LN_EPS);
      float mx = -3.0e38f;
#pragma unroll
      for (int v = 0; v < SEQ; ++v) {
        vals[v] = ((vals[v] - mu) * rstd * lng_s[v] + lnb_s[v]) * 0.125f;  // /sqrt(64)
        mx = fmaxf(mx, vals[v]);
      }
      float sum = 0.0f;
#pragma unroll
      for (int v = 0; v < SEQ; ++v) { vals[v] = __expf(vals[v] - mx); sum += vals[v]; }
      const float rs = 1.0f / sum;
#pragma unroll
      for (int v = 0; v < SEQ; ++v) S.as_[row][v] = (__bf16)(vals[v] * rs);
      S.as_[row][14] = (__bf16)0.0f;   // mask pad columns -> pad V rows never contribute
      S.as_[row][15] = (__bf16)0.0f;
    }

    // ---- out = att @ V (4 WMMA), stage C-layout into os, then TDM store ----
    v16bf aa = afrag_bf16(&S.as_[m16][0], half);
#pragma unroll
    for (int nt = 0; nt < 4; ++nt) {
      v16bf bvv = bfrag(&S.vt[nt * 16 + m16][0], half);
      v8f o = {};
      o = wmma_bf16(aa, bvv, o);
#pragma unroll
      for (int r = 0; r < 8; ++r) {
        const int mm = r + 8 * half;
        if (mm < SEQ) S.os[mm * EMB + nt * 16 + m16] = o[r];
      }
    }

    // fence DS stores so the TDM engine reads the finished tile, then DMA it out
    asm volatile("s_wait_dscnt 0" ::: "memory");
    {
      u32x4 g0; i32x8 g1;
      tdm_desc_1d((unsigned long long)(uintptr_t)(out + (size_t)tile * (SEQ * EMB)),
                  lds_off_of(&S.os[0]), SEQ * EMB, &g0, &g1);
      tdm_store(g0, g1);
    }
    // final store drains via S_ENDPGM's implicit wait-idle
  }
}

extern "C" void kernel_launch(void* const* d_in, const int* in_sizes, int n_in,
                              void* d_out, int out_size, void* d_ws, size_t ws_size,
                              hipStream_t stream) {
  const float* x   = (const float*)d_in[0];
  const float* w_c = (const float*)d_in[1];
  const float* Wq  = (const float*)d_in[2];
  const float* bq  = (const float*)d_in[3];
  const float* Wk  = (const float*)d_in[4];
  const float* bk  = (const float*)d_in[5];
  const float* Wv  = (const float*)d_in[6];
  const float* bv  = (const float*)d_in[7];
  const float* lng = (const float*)d_in[8];
  const float* lnb = (const float*)d_in[9];
  float* out = (float*)d_out;

  const int tiles  = BATCH * CHAN;            // 65536 independent (b,c) problems
  const int blocks = tiles / (WAVES * TPW);   // 2048 blocks x 8 waves x 4 tiles
  cmha_kernel<<<blocks, 256, 0, stream>>>(x, w_c, Wq, bq, Wk, bk, Wv, bv, lng, lnb, out);
}
